// GatedCrossAttention_12524124635286
// MI455X (gfx1250) — matrix operand order repaired and checked by reading, not compiled, not run on hardware
//
#include <hip/hip_runtime.h>
#include <hip/hip_bf16.h>

// GatedCrossAttention, MI455X (gfx1250).
// T=1 => softmax == 1 => q/k GEMMs dead; attn_out = ao[b] = Wo*v[b]+bo and
// gate constant c[b] = Wg2*ao[b]+bg are per-batch vectors.
// Real work: Y = X * Wg1^T (32768x1024x1024, bf16 WMMA, f32 accum) fused with
// sigmoid-gate, blend with x, and per-row LayerNorm.
//
// Wave layout (v2): 8 waves, wave w owns a UNIQUE 128-col slice (8 N-tiles)
// and BOTH 16-row halves (mt inner loop) => every B fragment is fetched from
// L2 exactly once per WG (2 GiB total instead of 4 GiB) and feeds 2 WMMAs.

typedef __attribute__((ext_vector_type(16))) __bf16 v16bf;
typedef __attribute__((ext_vector_type(8)))  float  v8f;
typedef unsigned short ushort_t;
typedef unsigned int   uint_t;

#define E_DIM 1024
#define S_DIM 4096
#define B_DIM 8
#define KD_DIM 256
#define BM 32           // rows per workgroup
#define BK 32           // k chunk (== WMMA K)
#define A_STRIDE 40     // ushorts per LDS A row (80B, bank-spread, 16B aligned)

union FragBF {
    v16bf bf;
    uint4 q[2];
};

__device__ __forceinline__ ushort_t f2bf(float f) {
    uint_t u = __float_as_uint(f);
    uint_t r = u + 0x7FFFu + ((u >> 16) & 1u);   // round-to-nearest-even
    return (ushort_t)(r >> 16);
}

// ---------------------------------------------------------------------------
// Kernel 0: per-batch constants  v[b] -> ao[b] -> c[b]
// ---------------------------------------------------------------------------
__global__ void gx_batch_consts(const float* __restrict__ spk,
                                const float* __restrict__ Wv,
                                const float* __restrict__ b_in,
                                const float* __restrict__ Wo,
                                const float* __restrict__ bo,
                                const float* __restrict__ Wg,
                                const float* __restrict__ bg,
                                float* __restrict__ ao,
                                float* __restrict__ cc) {
    __shared__ float vv[E_DIM];
    __shared__ float av[E_DIM];
    const int b = blockIdx.x;
    const int t = threadIdx.x;

    for (int f = t; f < E_DIM; f += 256) {
        float a = b_in[2 * E_DIM + f];
        const float* wr = Wv + (size_t)f * KD_DIM;
        const float* sp = spk + (size_t)b * KD_DIM;
        for (int e = 0; e < KD_DIM; ++e) a += sp[e] * wr[e];
        vv[f] = a;
    }
    __syncthreads();

    for (int f = t; f < E_DIM; f += 256) {
        float a = bo[f];
        const float* wr = Wo + (size_t)f * E_DIM;
        for (int e = 0; e < E_DIM; ++e) a += vv[e] * wr[e];
        av[f] = a;
        ao[(size_t)b * E_DIM + f] = a;
    }
    __syncthreads();

    for (int f = t; f < E_DIM; f += 256) {
        float a = bg[f];
        const float* wr = Wg + (size_t)f * (2 * E_DIM) + E_DIM;
        for (int e = 0; e < E_DIM; ++e) a += av[e] * wr[e];
        cc[(size_t)b * E_DIM + f] = a;
    }
}

// ---------------------------------------------------------------------------
// Kernel 1: Wg1 (= Wg[:, :E]) -> bf16 row-major (B-fragment friendly layout:
// column n of B == row n of Wg1; contiguous K runs per lane). 2 MiB -> L2.
// ---------------------------------------------------------------------------
__global__ void gx_cvt_wg1(const float* __restrict__ Wg,
                           ushort_t* __restrict__ Wb) {
    const int i4 = (blockIdx.x * 256 + threadIdx.x) * 4;
    const int n = i4 >> 10;
    const int k = i4 & (E_DIM - 1);
    const float4 v = *(const float4*)(Wg + (size_t)n * (2 * E_DIM) + k);
    uint2 p;
    p.x = (uint_t)f2bf(v.x) | ((uint_t)f2bf(v.y) << 16);
    p.y = (uint_t)f2bf(v.z) | ((uint_t)f2bf(v.w) << 16);
    *(uint2*)(Wb + i4) = p;
}

// ---------------------------------------------------------------------------
// Kernel 2: fused GEMM + gate + blend + LayerNorm.
// Grid: 1024 WGs x 256 threads (8 waves). WG: 32 rows x full N=1024.
// ---------------------------------------------------------------------------
__global__ void __launch_bounds__(256)
gx_main(const float* __restrict__ x,
        const ushort_t* __restrict__ Wb,
        const float* __restrict__ ao,
        const float* __restrict__ cc,
        const float* __restrict__ gamma,
        const float* __restrict__ beta,
        float* __restrict__ out) {
    __shared__ ushort_t As[BM * A_STRIDE];
    __shared__ float rsumS[BM][8];
    __shared__ float rsqS[BM][8];

    const int t    = threadIdx.x;
    const int gr0  = blockIdx.x * BM;        // first global row of this WG
    const int bb   = gr0 >> 12;              // batch (S=4096, 32 | 4096)
    const int lane = t & 31;
    const int w    = t >> 5;
    const int m    = lane & 15;              // N index within tile / A row sel
    const int hi   = lane >> 4;              // lane half (K split per ISA)
    const int ng   = w;                      // unique 128-col group per wave
    const int colb = ng << 7;

    // staging: thread t loads 4 consecutive k of one row
    const int srow = t >> 3;                 // 0..31
    const int scol = (t & 7) << 2;           // 0..28

    // acc[mt][nt]: 2 M-halves x 8 N-tiles = 128 VGPRs
    v8f acc[2][8];
    const v8f vz = {0.f, 0.f, 0.f, 0.f, 0.f, 0.f, 0.f, 0.f};
#pragma unroll
    for (int mt = 0; mt < 2; ++mt)
#pragma unroll
        for (int i = 0; i < 8; ++i) acc[mt][i] = vz;

    const ushort_t* ap0 = &As[(0 * 16 + m) * A_STRIDE + hi * 8];
    const ushort_t* ap1 = &As[(1 * 16 + m) * A_STRIDE + hi * 8];

    for (int kt = 0; kt < E_DIM / BK; ++kt) {
        // ---- stage 32x32 f32 -> bf16 into LDS ----
        const float4 xv = *(const float4*)(x + (size_t)(gr0 + srow) * E_DIM +
                                           kt * BK + scol);
        uint2 p;
        p.x = (uint_t)f2bf(xv.x) | ((uint_t)f2bf(xv.y) << 16);
        p.y = (uint_t)f2bf(xv.z) | ((uint_t)f2bf(xv.w) << 16);
        *(uint2*)&As[srow * A_STRIDE + scol] = p;
        __syncthreads();

        // ---- A fragments for both M-halves (ISA 7.12.2 16-bit 16x32) ----
        FragBF a0, a1;
        a0.q[0] = *(const uint4*)(ap0);
        a0.q[1] = *(const uint4*)(ap0 + 16);
        a1.q[0] = *(const uint4*)(ap1);
        a1.q[1] = *(const uint4*)(ap1 + 16);

        // ---- B fragments: one L2 fetch feeds two WMMAs ----
        const ushort_t* bp = Wb + (size_t)(colb + m) * E_DIM + kt * BK + hi * 8;
#pragma unroll
        for (int nt = 0; nt < 8; ++nt) {
            FragBF bfr;
            const ushort_t* bpp = bp + (size_t)nt * 16 * E_DIM;
            bfr.q[0] = *(const uint4*)(bpp);
            bfr.q[1] = *(const uint4*)(bpp + 16);
            acc[0][nt] = __builtin_amdgcn_wmma_f32_16x16x32_bf16(
                false, a0.bf, false, bfr.bf, (short)0, acc[0][nt], false, false);
            acc[1][nt] = __builtin_amdgcn_wmma_f32_16x16x32_bf16(
                false, a1.bf, false, bfr.bf, (short)0, acc[1][nt], false, false);
        }
        __syncthreads();
    }

    // ---- epilogue: y+c -> sigmoid -> blend with x -> row stats ----
    float psum[2][8], psq[2][8];
#pragma unroll
    for (int mt = 0; mt < 2; ++mt)
#pragma unroll
        for (int r = 0; r < 8; ++r) { psum[mt][r] = 0.f; psq[mt][r] = 0.f; }

#pragma unroll
    for (int nt = 0; nt < 8; ++nt) {
        const int col = colb + nt * 16 + m;
        const float cv = cc[(size_t)bb * E_DIM + col];
        const float av = ao[(size_t)bb * E_DIM + col];
#pragma unroll
        for (int mt = 0; mt < 2; ++mt) {
#pragma unroll
            for (int r = 0; r < 8; ++r) {
                const int grow = gr0 + mt * 16 + hi * 8 + r;  // C: VGPR r -> M=r(+8)
                const float y = acc[mt][nt][r] + cv;
                const float g = 1.f / (1.f + __expf(-y));
                const float xr = x[(size_t)grow * E_DIM + col];
                const float f = g * av + (1.f - g) * xr;
                acc[mt][nt][r] = f;
                psum[mt][r] += f;
                psq[mt][r]  += f * f;
            }
        }
    }

    // butterfly over the 16 lanes of each half (wave32)
#pragma unroll
    for (int mt = 0; mt < 2; ++mt) {
#pragma unroll
        for (int r = 0; r < 8; ++r) {
#pragma unroll
            for (int off = 1; off < 16; off <<= 1) {
                psum[mt][r] += __shfl_xor(psum[mt][r], off, 32);
                psq[mt][r]  += __shfl_xor(psq[mt][r], off, 32);
            }
        }
    }

    if (m < 8) {
#pragma unroll
        for (int mt = 0; mt < 2; ++mt) {
            float sv = 0.f, qv = 0.f;
#pragma unroll
            for (int r = 0; r < 8; ++r) {
                if (r == m) { sv = psum[mt][r]; qv = psq[mt][r]; }
            }
            const int rl = mt * 16 + hi * 8 + m;
            rsumS[rl][ng] = sv;
            rsqS[rl][ng]  = qv;
        }
    }
    __syncthreads();

    float mean[2][8], rstd[2][8];
#pragma unroll
    for (int mt = 0; mt < 2; ++mt) {
#pragma unroll
        for (int r = 0; r < 8; ++r) {
            const int rl = mt * 16 + hi * 8 + r;
            float s = 0.f, q = 0.f;
#pragma unroll
            for (int gi = 0; gi < 8; ++gi) { s += rsumS[rl][gi]; q += rsqS[rl][gi]; }
            const float mu = s * (1.f / (float)E_DIM);
            const float var = q * (1.f / (float)E_DIM) - mu * mu;
            mean[mt][r] = mu;
            rstd[mt][r] = rsqrtf(var + 1e-5f);
        }
    }

#pragma unroll
    for (int nt = 0; nt < 8; ++nt) {
        const int col = colb + nt * 16 + m;
        const float ga = gamma[col];
        const float be = beta[col];
#pragma unroll
        for (int mt = 0; mt < 2; ++mt) {
#pragma unroll
            for (int r = 0; r < 8; ++r) {
                const int grow = gr0 + mt * 16 + hi * 8 + r;
                out[(size_t)grow * E_DIM + col] =
                    (acc[mt][nt][r] - mean[mt][r]) * rstd[mt][r] * ga + be;
            }
        }
    }
}

// ---------------------------------------------------------------------------
extern "C" void kernel_launch(void* const* d_in, const int* in_sizes, int n_in,
                              void* d_out, int out_size, void* d_ws, size_t ws_size,
                              hipStream_t stream) {
    const float* x    = (const float*)d_in[0];   // (B,S,E)
    const float* spk  = (const float*)d_in[1];   // (B,KD)
    // d_in[2] Wq, d_in[3] Wk : dead (softmax over length-1 axis == 1)
    const float* Wv   = (const float*)d_in[4];   // (E,KD)
    const float* b_in = (const float*)d_in[5];   // (3E,)
    const float* Wo   = (const float*)d_in[6];   // (E,E)
    const float* bo   = (const float*)d_in[7];   // (E,)
    const float* Wg   = (const float*)d_in[8];   // (E,2E)
    const float* bg   = (const float*)d_in[9];   // (E,)
    const float* gam  = (const float*)d_in[10];  // (E,)
    const float* bet  = (const float*)d_in[11];  // (E,)
    float* out = (float*)d_out;

    // workspace: [0, 2MiB) Wg1 bf16 ; then ao (32KB) ; then c (32KB)
    ushort_t* Wb = (ushort_t*)d_ws;
    float* ao = (float*)((char*)d_ws + (2u << 20));
    float* cc = ao + B_DIM * E_DIM;

    gx_cvt_wg1<<<1024, 256, 0, stream>>>(Wg, Wb);
    gx_batch_consts<<<B_DIM, 256, 0, stream>>>(spk, Wv, b_in, Wo, bo, Wg, bg, ao, cc);
    gx_main<<<(B_DIM * S_DIM) / BM, 256, 0, stream>>>(x, Wb, ao, cc, gam, bet, out);
}
